// Encoder_47098611368204
// MI455X (gfx1250) — matrix operand order repaired and checked by reading, not compiled, not run on hardware
//
#include <hip/hip_runtime.h>

typedef __attribute__((ext_vector_type(16))) _Float16 v16h;
typedef __attribute__((ext_vector_type(8)))  float    v8f;
typedef __attribute__((ext_vector_type(4)))  unsigned int u32x4;

#define NEG_SLOPE 0.2f

union AFrag { v16h h; u32x4 q[2]; };
union BFrag { v16h h; u32x4 q[2]; };
union HVec  { _Float16 e[8]; u32x4 q; };

// ---------------------------------------------------------------------------
// Weight packing: OIHW f32 -> [Cout][Kpad] f16, K order = tap-major/channel-minor
//   k = tap*Ctot + c   (tap = ky*3+kx), zero padded to Kpad (multiple of 32).
// ---------------------------------------------------------------------------
__global__ void pack_w_kernel(const float* __restrict__ w, _Float16* __restrict__ dst,
                              int Cout, int Ctot, int Kpad) {
  int i = blockIdx.x * 256 + threadIdx.x;
  if (i >= Cout * Kpad) return;
  int o = i / Kpad;
  int k = i - o * Kpad;
  const int Kreal = Ctot * 9;
  _Float16 v = (_Float16)0.0f;
  if (k < Kreal) {
    int tap = k / Ctot;
    int c   = k - tap * Ctot;
    v = (_Float16)w[(o * Ctot + c) * 9 + tap];
  }
  dst[i] = v;
}

__global__ void fill_zero_h16(_Float16* __restrict__ p, long n) {
  long i = (long)blockIdx.x * 256 + threadIdx.x;
  if (i < n) p[i] = (_Float16)0.0f;
}

// ---------------------------------------------------------------------------
// Front conv of stage 1: 1->16 channels, stride 2, 128->64 (0.05% of FLOPs,
// C=1 is too skinny for the matrix path) -> direct conv, one thread per pixel.
// Input x is (B=8,T=10,1,128,128) f32; output feat is ((t*8+b),64,64,16) f16.
// ---------------------------------------------------------------------------
__global__ void front1_direct_kernel(const float* __restrict__ x,
                                     const float* __restrict__ W1,
                                     const float* __restrict__ b1,
                                     _Float16* __restrict__ feat) {
  int idx = blockIdx.x * 256 + threadIdx.x;
  if (idx >= 80 * 4096) return;
  int img = idx >> 12, pix = idx & 4095;
  int oy = pix >> 6, ox = pix & 63;
  int t = img >> 3, b = img & 7;
  const float* xi = x + ((size_t)b * 10 + t) * 16384;
  float v[9];
#pragma unroll
  for (int tap = 0; tap < 9; ++tap) {
    int ky = tap / 3, kx = tap - 3 * (tap / 3);
    int iy = 2 * oy - 1 + ky, ix = 2 * ox - 1 + kx;
    v[tap] = ((unsigned)iy < 128u && (unsigned)ix < 128u) ? xi[iy * 128 + ix] : 0.0f;
  }
  union { _Float16 e[16]; u32x4 q[2]; } O;
#pragma unroll
  for (int c = 0; c < 16; ++c) {
    float s = b1[c];
#pragma unroll
    for (int tap = 0; tap < 9; ++tap) s += W1[c * 9 + tap] * v[tap];
    O.e[c] = (_Float16)((s > 0.0f) ? s : NEG_SLOPE * s);
  }
  u32x4* dst = (u32x4*)(feat + (size_t)idx * 16);
  dst[0] = O.q[0];
  dst[1] = O.q[1];
}

// ---------------------------------------------------------------------------
// Implicit-GEMM 3x3 conv (pad=1) on WMMA, NHWC activations, tap-major K.
// blockIdx.y = M-tile (no division); all spatial decode is shift/mask since
// every H/W in this net is a power of two. One wave32 owns a 16x64 tile:
// 4 accumulators share one A fragment per K-step; all operand loads are b128.
// MODE 0: front  -> out16 = f16(leaky_relu(acc+bias))
// MODE 1: gates  -> g=sigmoid(acc+bias); c<hid: rh=g*h (f16)  else z16=g
// MODE 2: cand   -> hn = z*h + (1-z)*tanh(acc+bias); h16/out16 = f16(hn);
//                   out_f32 (if set) gets hn scattered to NCHW f32.
// ---------------------------------------------------------------------------
template <int MODE>
__global__ void __launch_bounds__(256)
conv_wmma_nhwc(const _Float16* __restrict__ in0, int C0,
               const _Float16* __restrict__ in1, int C1,
               const _Float16* __restrict__ wpk, int Kpad, int Kreal, int Ctot,
               const float* __restrict__ bias,
               int lgHWo, int lgWo, int lgWi, int Hi, int Wi,
               int stride, int Cout, int nGroups,
               _Float16* __restrict__ out16,
               _Float16* __restrict__ z16,
               _Float16* __restrict__ h16,
               float* __restrict__ out_f32) {
  const int lane = threadIdx.x & 31;
  const int wave = threadIdx.x >> 5;

  const int ng = blockIdx.x * 8 + wave;
  if (ng >= nGroups) return;                          // wave-uniform
  const int mtile = blockIdx.y;

  const int HWo = 1 << lgHWo;
  const int m0    = mtile << 4;
  const int laneM = lane & 15;
  const int laneH = lane >> 4;                        // 0 or 1 (K half)
  const _Float16* aptr = wpk + (size_t)(m0 + laneM) * Kpad + (laneH << 4);
  const int kbB = laneH << 4;

  const int HWi = Hi * Wi;
  int imgI[4], pixI[4], oyS[4], oxS[4];
  const _Float16* base0[4];
  const _Float16* base1[4];
#pragma unroll
  for (int i = 0; i < 4; ++i) {
    int n   = ((ng << 2) + i) * 16 + laneM;
    int img = n >> lgHWo;
    int pix = n & (HWo - 1);
    int oy = pix >> lgWo, ox = pix & ((1 << lgWo) - 1);
    imgI[i] = img;
    pixI[i] = pix;
    oyS[i]  = oy * stride - 1;
    oxS[i]  = ox * stride - 1;
    base0[i] = in0 + (size_t)img * HWi * C0;
    base1[i] = in1 ? (in1 + (size_t)img * HWi * C1) : (const _Float16*)0;
  }

  v8f acc[4] = {};
  const u32x4 zero = {0u, 0u, 0u, 0u};

  // incremental tap/channel decode for this lane-half's 16-group
  int within = kbB, ky = 0, kx = 0;
  const int ksteps = Kpad >> 5;
#pragma unroll 2
  for (int ks = 0; ks < ksteps; ++ks) {
    AFrag A;
    A.q[0] = *(const u32x4*)(aptr + (ks << 5));
    A.q[1] = *(const u32x4*)(aptr + (ks << 5) + 8);

    const int  kb     = (ks << 5) + kbB;
    const bool gvalid = (kb < Kreal);
    const bool use1   = (within >= C0);
    const int  c0     = use1 ? (within - C0) : within;
    const int  Cl     = use1 ? C1 : C0;

    BFrag Bv[4];
#pragma unroll
    for (int i = 0; i < 4; ++i) {
      const int iy = oyS[i] + ky;
      const int ix = oxS[i] + kx;
      const bool ok = gvalid && ((unsigned)iy < (unsigned)Hi) && ((unsigned)ix < (unsigned)Wi);
      if (ok) {
        const _Float16* src =
            (use1 ? base1[i] : base0[i]) + ((size_t)((iy << lgWi) + ix)) * Cl + c0;
        Bv[i].q[0] = *(const u32x4*)src;
        Bv[i].q[1] = *(const u32x4*)(src + 8);
      } else {
        Bv[i].q[0] = zero;
        Bv[i].q[1] = zero;
      }
    }

    acc[0] = __builtin_amdgcn_wmma_f32_16x16x32_f16(false, A.h, false, Bv[0].h, (short)0, acc[0], false, false);
    acc[1] = __builtin_amdgcn_wmma_f32_16x16x32_f16(false, A.h, false, Bv[1].h, (short)0, acc[1], false, false);
    acc[2] = __builtin_amdgcn_wmma_f32_16x16x32_f16(false, A.h, false, Bv[2].h, (short)0, acc[2], false, false);
    acc[3] = __builtin_amdgcn_wmma_f32_16x16x32_f16(false, A.h, false, Bv[3].h, (short)0, acc[3], false, false);

    within += 32;
    if (within >= Ctot) {
      within -= Ctot;
      ++kx;
      if (kx == 3) { kx = 0; ++ky; }
    }
  }

  // ---- Epilogue: D tile lane (laneM)=N column, VGPR j -> channel cbase+j ----
  const int cbase = m0 + (laneH << 3);
  float bb[8];
#pragma unroll
  for (int j = 0; j < 8; ++j) bb[j] = bias[cbase + j];

#pragma unroll
  for (int i = 0; i < 4; ++i) {
    const size_t nb = ((size_t)imgI[i] << lgHWo) + pixI[i];
    if (MODE == 0) {
      HVec O;
#pragma unroll
      for (int j = 0; j < 8; ++j) {
        const float y = acc[i][j] + bb[j];
        O.e[j] = (_Float16)((y > 0.0f) ? y : NEG_SLOPE * y);
      }
      *(u32x4*)(out16 + nb * Cout + cbase) = O.q;
    } else if (MODE == 1) {
      const int hid = Cout >> 1;
      HVec O;
      if (cbase < hid) {                 // r-gate block: emit r*h for cand conv
        const size_t id = nb * hid + cbase;
        HVec Hv;
        Hv.q = *(const u32x4*)(h16 + id);
#pragma unroll
        for (int j = 0; j < 8; ++j) {
          const float g = 1.0f / (1.0f + __expf(-(acc[i][j] + bb[j])));
          O.e[j] = (_Float16)(g * (float)Hv.e[j]);
        }
        *(u32x4*)(out16 + id) = O.q;
      } else {                           // z-gate block
        const size_t id = nb * hid + (cbase - hid);
#pragma unroll
        for (int j = 0; j < 8; ++j) {
          const float g = 1.0f / (1.0f + __expf(-(acc[i][j] + bb[j])));
          O.e[j] = (_Float16)g;
        }
        *(u32x4*)(z16 + id) = O.q;
      }
    } else {                             // MODE 2: cand + GRU state update
      const size_t id = nb * Cout + cbase;
      HVec Hv, Zv, O;
      Hv.q = *(const u32x4*)(h16 + id);
      Zv.q = *(const u32x4*)(z16 + id);
#pragma unroll
      for (int j = 0; j < 8; ++j) {
        const float z  = (float)Zv.e[j];
        const float hp = (float)Hv.e[j];
        const float cd = tanhf(acc[i][j] + bb[j]);
        const float hn = z * hp + (1.0f - z) * cd;
        O.e[j] = (_Float16)hn;
        if (out_f32)
          out_f32[(((size_t)imgI[i] * Cout + (cbase + j)) << lgHWo) + pixI[i]] = hn;
      }
      *(u32x4*)(h16 + id) = O.q;
      *(u32x4*)(out16 + id) = O.q;
    }
  }
}

// ---------------------------------------------------------------------------
// Host orchestration
// ---------------------------------------------------------------------------
extern "C" void kernel_launch(void* const* d_in, const int* in_sizes, int n_in,
                              void* d_out, int out_size, void* d_ws, size_t ws_size,
                              hipStream_t stream) {
  (void)in_sizes; (void)n_in; (void)out_size; (void)ws_size;

  const float* x   = (const float*)d_in[0];
  const float* W1  = (const float*)d_in[1];  const float* b1  = (const float*)d_in[2];
  const float* Wg1 = (const float*)d_in[3];  const float* bg1 = (const float*)d_in[4];
  const float* Wc1 = (const float*)d_in[5];  const float* bc1 = (const float*)d_in[6];
  const float* W2  = (const float*)d_in[7];  const float* b2  = (const float*)d_in[8];
  const float* Wg2 = (const float*)d_in[9];  const float* bg2 = (const float*)d_in[10];
  const float* Wc2 = (const float*)d_in[11]; const float* bc2 = (const float*)d_in[12];
  const float* W3  = (const float*)d_in[13]; const float* b3  = (const float*)d_in[14];
  const float* Wg3 = (const float*)d_in[15]; const float* bg3 = (const float*)d_in[16];
  const float* Wc3 = (const float*)d_in[17]; const float* bc3 = (const float*)d_in[18];

  char*  base = (char*)d_ws;
  size_t cur  = 0;
  auto allocH = [&](size_t nelem) -> _Float16* {
    _Float16* p = (_Float16*)(base + cur);
    cur += ((nelem * sizeof(_Float16) + 255) & ~(size_t)255);
    return p;
  };

  // packed f16 weights [Cout][Kpad], tap-major K
  _Float16* w_g1 = allocH((size_t)128 * 736);
  _Float16* w_c1 = allocH((size_t)64  * 736);
  _Float16* w_f2 = allocH((size_t)64  * 576);
  _Float16* w_g2 = allocH((size_t)192 * 1440);
  _Float16* w_c2 = allocH((size_t)96  * 1440);
  _Float16* w_f3 = allocH((size_t)96  * 864);
  _Float16* w_g3 = allocH((size_t)192 * 1728);
  _Float16* w_c3 = allocH((size_t)96  * 1728);
  // shared NHWC feature buffer (max over stages)
  _Float16* feat = allocH(5242880);      // 80*4096*16 == 80*1024*64
  // shared per-timestep GRU outputs (stage1 size covers stage2)
  _Float16* outs = allocH(20971520);     // 80*4096*64
  // per-stage GRU state (NHWC f16)
  _Float16* h1  = allocH(2097152); _Float16* rh1 = allocH(2097152); _Float16* z1 = allocH(2097152);
  _Float16* h2  = allocH(786432);  _Float16* rh2 = allocH(786432);  _Float16* z2 = allocH(786432);
  _Float16* h3  = allocH(196608);  _Float16* rh3 = allocH(196608);  _Float16* z3 = allocH(196608);
  _Float16* dump= allocH(196608);  // stage3 per-t output sink (never read)

  auto pack = [&](const float* w, _Float16* dst, int Cout, int Ctot, int Kpad) {
    int total = Cout * Kpad;
    pack_w_kernel<<<(total + 255) / 256, 256, 0, stream>>>(w, dst, Cout, Ctot, Kpad);
  };
  auto fill = [&](_Float16* p, long nelem) {
    fill_zero_h16<<<(int)((nelem + 255) / 256), 256, 0, stream>>>(p, nelem);
  };
  // N_img, out lg sizes, in sizes, etc. -> launch one conv
  auto conv = [&](int mode,
                  const _Float16* in0, int C0, const _Float16* in1, int C1,
                  const _Float16* wpk, int Kpad, int Kreal, int Ctot, const float* bias,
                  int N_img, int lgHWo, int lgWo, int lgWi, int Hi, int Wi,
                  int stride, int Cout,
                  _Float16* o16, _Float16* z, _Float16* h, float* of32) {
    const int nGroups = (N_img << lgHWo) >> 6;   // (N_img*HWo/16)/4
    dim3 grid((nGroups + 7) / 8, Cout >> 4);
    if (mode == 0)
      conv_wmma_nhwc<0><<<grid, 256, 0, stream>>>(in0, C0, in1, C1, wpk, Kpad, Kreal, Ctot,
          bias, lgHWo, lgWo, lgWi, Hi, Wi, stride, Cout, nGroups, o16, z, h, of32);
    else if (mode == 1)
      conv_wmma_nhwc<1><<<grid, 256, 0, stream>>>(in0, C0, in1, C1, wpk, Kpad, Kreal, Ctot,
          bias, lgHWo, lgWo, lgWi, Hi, Wi, stride, Cout, nGroups, o16, z, h, of32);
    else
      conv_wmma_nhwc<2><<<grid, 256, 0, stream>>>(in0, C0, in1, C1, wpk, Kpad, Kreal, Ctot,
          bias, lgHWo, lgWo, lgWi, Hi, Wi, stride, Cout, nGroups, o16, z, h, of32);
  };

  // ---- preprocessing ----
  pack(Wg1, w_g1, 128, 80,  736);
  pack(Wc1, w_c1, 64,  80,  736);
  pack(W2,  w_f2, 64,  64,  576);
  pack(Wg2, w_g2, 192, 160, 1440);
  pack(Wc2, w_c2, 96,  160, 1440);
  pack(W3,  w_f3, 96,  96,  864);
  pack(Wg3, w_g3, 192, 192, 1728);
  pack(Wc3, w_c3, 96,  192, 1728);

  float* out_h1 = (float*)d_out;        // (8,64,64,64) NCHW f32
  float* out_h2 = out_h1 + 2097152;     // (8,96,32,32)
  float* out_h3 = out_h2 + 786432;      // (8,96,16,16)

  // ---- Stage 1: front conv 1->16 s2 (direct); GRU in=16 hid=64 @64x64 ----
  front1_direct_kernel<<<(80 * 4096 + 255) / 256, 256, 0, stream>>>(x, W1, b1, feat);
  fill(h1, 2097152);
  for (int t = 0; t < 10; ++t) {
    const _Float16* ft = feat + (size_t)t * 8 * 4096 * 16;
    conv(1, ft, 16, h1, 64, w_g1, 736, 720, 80, bg1,
         8, 12, 6, 6, 64, 64, 1, 128, rh1, z1, h1, nullptr);
    conv(2, ft, 16, rh1, 64, w_c1, 736, 720, 80, bc1,
         8, 12, 6, 6, 64, 64, 1, 64,
         outs + (size_t)t * 8 * 4096 * 64, z1, h1, (t == 9) ? out_h1 : nullptr);
  }

  // ---- Stage 2: front conv 64->64 s2 (64->32); GRU in=64 hid=96 @32x32 ----
  conv(0, outs, 64, nullptr, 0, w_f2, 576, 576, 64, b2,
       80, 10, 5, 6, 64, 64, 2, 64, feat, nullptr, nullptr, nullptr);
  fill(h2, 786432);
  for (int t = 0; t < 10; ++t) {
    const _Float16* ft = feat + (size_t)t * 8 * 1024 * 64;
    conv(1, ft, 64, h2, 96, w_g2, 1440, 1440, 160, bg2,
         8, 10, 5, 5, 32, 32, 1, 192, rh2, z2, h2, nullptr);
    conv(2, ft, 64, rh2, 96, w_c2, 1440, 1440, 160, bc2,
         8, 10, 5, 5, 32, 32, 1, 96,
         outs + (size_t)t * 8 * 1024 * 96, z2, h2, (t == 9) ? out_h2 : nullptr);
  }

  // ---- Stage 3: front conv 96->96 s2 (32->16); GRU in=96 hid=96 @16x16 ----
  conv(0, outs, 96, nullptr, 0, w_f3, 864, 864, 96, b3,
       80, 8, 4, 5, 32, 32, 2, 96, feat, nullptr, nullptr, nullptr);
  fill(h3, 196608);
  for (int t = 0; t < 10; ++t) {
    const _Float16* ft = feat + (size_t)t * 8 * 256 * 96;
    conv(1, ft, 96, h3, 96, w_g3, 1728, 1728, 192, bg3,
         8, 8, 4, 4, 16, 16, 1, 192, rh3, z3, h3, nullptr);
    conv(2, ft, 96, rh3, 96, w_c3, 1728, 1728, 192, bc3,
         8, 8, 4, 4, 16, 16, 1, 96,
         dump, z3, h3, (t == 9) ? out_h3 : nullptr);
  }
}
